// RandomizedLongformerSelfAttention_45509473468724
// MI455X (gfx1250) — compile-verified, matrix operand order checked by reading, not run on hardware
//
#include <hip/hip_runtime.h>
#include <hip/hip_bf16.h>

// ---------------------------------------------------------------------------
// Longformer self-attention for MI455X (gfx1250), wave32 + WMMA bf16 + TDM.
//   B=1, S=4096, D=1024, H=16, d=64, W=256, C=16 chunks, context=3W=768,
//   band width 2W+1=513.
// Pipeline:
//   1) cvt_kernel : fp32 -> bf16 for hidden_states and Wq/Wk/Wv
//   2) proj_kernel: QKV GEMMs with v_wmma_f32_16x16x32_bf16,
//                   writes Q[h][s][d] (pre-scaled by 1/8), K[h][s][d], Vt[h][d][s]
//   3) attn_kernel: per (h, chunk, 32-query slab):
//        - TENSOR_LOAD_TO_LDS prefetches the 64x768 V^T context tile (overlaps
//          with score/softmax compute, waited with s_wait_tensorcnt)
//        - scores (WMMA) -> banded mask -> softmax -> P.V (WMMA, B from LDS)
// ---------------------------------------------------------------------------

typedef __attribute__((ext_vector_type(16))) __bf16 v16bf;
typedef __attribute__((ext_vector_type(8)))  __bf16 v8bf;
typedef __attribute__((ext_vector_type(8)))  float  v8f;
typedef __attribute__((ext_vector_type(4)))  unsigned int u32x4;
typedef __attribute__((ext_vector_type(8)))  unsigned int u32x8;

#define S_LEN  4096
#define DMODEL 1024
#define NHEAD  16
#define HDIM   64
#define CTX    768   // 3*W context columns per chunk

static __device__ __forceinline__ v16bf cat8(v8bf lo, v8bf hi) {
  return __builtin_shufflevector(lo, hi, 0,1,2,3,4,5,6,7,8,9,10,11,12,13,14,15);
}

// Row-major 16x32 bf16 operand fetch matching the CDNA5 WMMA A/B per-lane
// layout: lane (mod 16) = row; VGPR0-3 hold K=kb..kb+7, VGPR4-7 K=kb+16..kb+23
// with kb = k + (lane>=16 ? 8 : 0).  Caller passes kb already including the
// lane-half offset.
static __device__ __forceinline__ v16bf load_tile16(const __bf16* base, int stride,
                                                    int row, int kb) {
  const __bf16* p = base + (size_t)row * stride + kb;
  v8bf lo = *reinterpret_cast<const v8bf*>(p);
  v8bf hi = *reinterpret_cast<const v8bf*>(p + 16);
  return cat8(lo, hi);
}

static __device__ __forceinline__ v8f wmma_bf16(v16bf a, v16bf b, v8f c) {
  return __builtin_amdgcn_wmma_f32_16x16x32_bf16(false, a, false, b,
                                                 (short)0, c, false, false);
}

static __device__ __forceinline__ float wave_max(float v) {
  #pragma unroll
  for (int o = 16; o > 0; o >>= 1) v = fmaxf(v, __shfl_xor(v, o, 32));
  return v;
}
static __device__ __forceinline__ float wave_sum(float v) {
  #pragma unroll
  for (int o = 16; o > 0; o >>= 1) v += __shfl_xor(v, o, 32);
  return v;
}

// ---------------------------------------------------------------------------
// Kernel 1: fp32 -> bf16 conversion (X and the three weight matrices)
// ---------------------------------------------------------------------------
__global__ __launch_bounds__(256) void cvt_kernel(
    const float* __restrict__ X,  const float* __restrict__ Wq,
    const float* __restrict__ Wk, const float* __restrict__ Wv,
    __bf16* __restrict__ Xb,  __bf16* __restrict__ Wqb,
    __bf16* __restrict__ Wkb, __bf16* __restrict__ Wvb) {
  const int t = blockIdx.x * 256 + threadIdx.x;          // 0 .. 2^20-1
  Wqb[t] = (__bf16)Wq[t];
  Wkb[t] = (__bf16)Wk[t];
  Wvb[t] = (__bf16)Wv[t];
  #pragma unroll
  for (int j = 0; j < 4; ++j) {                          // X has 4*2^20 elems
    const int i = j * (1 << 20) + t;
    Xb[i] = (__bf16)X[i];
  }
}

// ---------------------------------------------------------------------------
// Kernel 2: QKV projection GEMM.  Each wave computes a 32(M) x 64(N) tile of
// one of Q/K/V with K-dim 1024.  out[s][o] = sum_k X[s][k]*W[o][k] + bias[o].
// jobs = 3 mats * 128 s-groups * 16 o-groups = 6144 ; 8 waves/block -> 768 blk
// ---------------------------------------------------------------------------
__global__ __launch_bounds__(256) void proj_kernel(
    const __bf16* __restrict__ Xb,  const __bf16* __restrict__ Wqb,
    const __bf16* __restrict__ Wkb, const __bf16* __restrict__ Wvb,
    const float* __restrict__ bq, const float* __restrict__ bk,
    const float* __restrict__ bv,
    __bf16* __restrict__ Qb, __bf16* __restrict__ Kb, __bf16* __restrict__ Vtb) {
  const int lane   = threadIdx.x & 31;
  const int wave   = threadIdx.x >> 5;
  const int laneLo = lane & 15;
  const int laneHi = lane >> 4;

  const int job    = blockIdx.x * 8 + wave;
  const int matsel = job >> 11;          // /2048 : 0=Q 1=K 2=V
  const int rem    = job & 2047;
  const int s0     = (rem >> 4) * 32;
  const int og     = rem & 15;
  const int o0     = og * 64;

  const __bf16* Wb   = (matsel == 0) ? Wqb : (matsel == 1) ? Wkb : Wvb;
  const float*  bias = (matsel == 0) ? bq  : (matsel == 1) ? bk  : bv;

  v8f acc[2][4] = {};

  for (int k = 0; k < DMODEL; k += 32) {
    const int kb = k + laneHi * 8;
    if (k + 32 < DMODEL) {
      __builtin_prefetch(Xb + (size_t)(s0 + laneLo) * DMODEL + kb + 32, 0, 1);
    }
    v16bf a0 = load_tile16(Xb, DMODEL, s0 + laneLo,      kb);
    v16bf a1 = load_tile16(Xb, DMODEL, s0 + 16 + laneLo, kb);
    #pragma unroll
    for (int nt = 0; nt < 4; ++nt) {
      v16bf b = load_tile16(Wb, DMODEL, o0 + nt * 16 + laneLo, kb);
      acc[0][nt] = wmma_bf16(a0, b, acc[0][nt]);
      acc[1][nt] = wmma_bf16(a1, b, acc[1][nt]);
    }
  }

  const int h = og;                       // o0 = h*64
  #pragma unroll
  for (int mt = 0; mt < 2; ++mt) {
    #pragma unroll
    for (int nt = 0; nt < 4; ++nt) {
      #pragma unroll
      for (int i = 0; i < 8; ++i) {
        const int srow = s0 + mt * 16 + laneHi * 8 + i;   // C/D layout: M = i (+8 hi)
        const int o    = o0 + nt * 16 + laneLo;
        const int dd   = o & 63;
        const float v  = acc[mt][nt][i] + bias[o];
        if (matsel == 0) {
          Qb[((size_t)h * S_LEN + srow) * HDIM + dd] = (__bf16)(v * 0.125f);
        } else if (matsel == 1) {
          Kb[((size_t)h * S_LEN + srow) * HDIM + dd] = (__bf16)v;
        } else {
          Vtb[((size_t)h * HDIM + dd) * S_LEN + srow] = (__bf16)v;  // transposed
        }
      }
    }
  }
}

// ---------------------------------------------------------------------------
// Kernel 3: banded attention.  Block = (h, chunk c, 32-query slab qb).
// grid = 16*16*8 = 2048 blocks of 256 threads (8 waves).
// LDS: scores 96KB + probs 48KB + V-stage 96KB = 240KB (<320KB WGP budget).
// ---------------------------------------------------------------------------
__global__ __launch_bounds__(256) void attn_kernel(
    const __bf16* __restrict__ Qb, const __bf16* __restrict__ Kb,
    const __bf16* __restrict__ Vtb,
    const float* __restrict__ amask, const int* __restrict__ idxmask,
    float* __restrict__ out) {
  __shared__ __attribute__((aligned(16))) float  Sld[32 * CTX];       // 96 KB
  __shared__ __attribute__((aligned(16))) __bf16 Pld[32 * CTX];       // 48 KB
  __shared__ __attribute__((aligned(16))) __bf16 Vstage[HDIM * CTX];  // 96 KB

  const int lane   = threadIdx.x & 31;
  const int wave   = threadIdx.x >> 5;
  const int laneLo = lane & 15;
  const int laneHi = lane >> 4;

  const int h     = blockIdx.x >> 7;
  const int rem   = blockIdx.x & 127;
  const int c     = rem >> 3;
  const int qb    = rem & 7;
  const int qg0   = c * 256 + qb * 32;    // first global query of this slab
  const int cbase = c * 256 - 256;        // global pos of context column 0
  // Clamp the staged context window fully in-bounds; edge chunks get a shifted
  // window and phase-3 re-shifts (affected P entries are exactly 0).
  const int start = min(max(cbase, 0), S_LEN - CTX);
  const int shift = cbase - start;        // -256, 0, or +256

  const __bf16* Qh = Qb  + (size_t)h * S_LEN * HDIM;
  const __bf16* Kh = Kb  + (size_t)h * S_LEN * HDIM;
  const __bf16* Vh = Vtb + (size_t)h * HDIM * S_LEN;

  // ---- Phase 0: kick off the TDM load of the V^T context tile -------------
  // 2D tile: tile_dim0 = 768 contiguous bf16 (s), tile_dim1 = 64 rows (d),
  // row stride 4096 elements.  Issued by wave 0 only (TDM ignores EXEC);
  // overlaps with phases 1-2, waited before phase 3.
  if (wave == 0) {
    const unsigned long long ga = (unsigned long long)(uintptr_t)(Vh + start);
    u32x4 g0;
    g0[0] = 1u;                                        // count=1, no gather
    g0[1] = (unsigned)(uintptr_t)(&Vstage[0]);         // lds_addr (byte offset)
    g0[2] = (unsigned)ga;                              // global_addr[31:0]
    g0[3] = (unsigned)((ga >> 32) & 0x1FFFFFFu)        // global_addr[56:32]
          | (2u << 30);                                // type=2 ("image")
    u32x8 g1;
    g1[0] = 1u << 16;                                  // data_size=1 (2 bytes)
    g1[1] = (unsigned)(S_LEN & 0xFFFF) << 16;          // tensor_dim0 lo16
    g1[2] = (unsigned)(S_LEN >> 16) | ((unsigned)HDIM << 16); // dim0 hi | dim1 lo
    g1[3] = (unsigned)CTX << 16;                       // dim1 hi(0) | tile_dim0
    g1[4] = (unsigned)HDIM;                            // tile_dim1 | tile_dim2=0
    g1[5] = (unsigned)S_LEN;                           // tensor_dim0_stride lo32
    g1[6] = 0u;                                        // stride hi | dim1_stride lo
    g1[7] = 0u;                                        // dim1_stride hi
    asm volatile("tensor_load_to_lds %0, %1" :: "s"(g0), "s"(g1) : "memory");
  }

  // ---- Phase 1: scores S = Q.K^T (masked) into LDS ------------------------
  {
    const int mt   = wave & 1;            // which 16-row M tile
    const int ntb  = wave >> 1;           // 4 waves split the 48 N tiles
    const int qrow = qg0 + mt * 16 + laneLo;
    const v16bf qa0 = load_tile16(Qh, HDIM, qrow, laneHi * 8);
    const v16bf qa1 = load_tile16(Qh, HDIM, qrow, 32 + laneHi * 8);

    for (int j = 0; j < 12; ++j) {
      const int nt  = ntb + 4 * j;
      const int y   = nt * 16 + laneLo;   // context column for this lane
      const int kg  = cbase + y;          // global key position
      const int kgc = min(max(kg, 0), S_LEN - 1);
      v16bf kb0 = load_tile16(Kh, HDIM, kgc, laneHi * 8);
      v16bf kb1 = load_tile16(Kh, HDIM, kgc, 32 + laneHi * 8);
      v8f z = {};
      z = wmma_bf16(qa0, kb0, z);
      z = wmma_bf16(qa1, kb1, z);

      const bool  vkg = (kg >= 0) && (kg < S_LEN);
      const float fmk = vkg ? ((amask[kgc] != 0.0f) ? -10000.0f : 0.0f) : 0.0f;
      #pragma unroll
      for (int i = 0; i < 8; ++i) {
        const int m = laneHi * 8 + i;
        const int x = qb * 32 + mt * 16 + m;          // query idx within chunk
        const bool band = (y >= x) && (y <= x + 512); // 2W+1 = 513 diagonal band
        const float sc = (band && vkg) ? (z[i] + fmk) : -1e30f;
        Sld[(mt * 16 + m) * CTX + y] = sc;
      }
    }
  }
  __syncthreads();

  // ---- Phase 2: row softmax over the 768 context columns ------------------
  {
    const int r0 = wave * 4;
    for (int ri = 0; ri < 4; ++ri) {
      const int r = r0 + ri;
      float* srow = &Sld[r * CTX];
      float mx = -1e30f;
      for (int y = lane; y < CTX; y += 32) mx = fmaxf(mx, srow[y]);
      mx = wave_max(mx);
      float sum = 0.0f;
      for (int y = lane; y < CTX; y += 32) {
        const float e = __expf(srow[y] - mx);
        srow[y] = e;
        sum += e;
      }
      sum = wave_sum(sum);
      float inv = 1.0f / sum;
      if (idxmask[qg0 + r] != 0) inv = 0.0f;          // is_index_masked
      __bf16* prow = &Pld[r * CTX];
      for (int y = lane; y < CTX; y += 32) prow[y] = (__bf16)(srow[y] * inv);
    }
  }
  // Drain the tensor DMA (wave 0 issued it; others have TENSORcnt==0), then
  // make the staged tile visible to the whole workgroup.
  if (wave == 0) __builtin_amdgcn_s_wait_tensorcnt(0);
  __syncthreads();

  // ---- Phase 3: O = P.V  (32x768 @ 768x64), one 16x16 tile per wave -------
  // A from Pld, B from the TDM-staged V^T tile in LDS.
  {
    const int mt = wave & 1;
    const int nt = wave >> 1;             // 0..3 -> 16 d-columns each
    const int prow = mt * 16 + laneLo;
    const int drow = nt * 16 + laneLo;
    const __bf16* vrow = &Vstage[(size_t)drow * CTX];
    v8f o = {};
    for (int kk = 0; kk < 24; ++kk) {
      const int kb = kk * 32 + laneHi * 8;
      const __bf16* pp = &Pld[prow * CTX + kb];
      v16bf a = cat8(*reinterpret_cast<const v8bf*>(pp),
                     *reinterpret_cast<const v8bf*>(pp + 16));
      // B operand: rows of V^T, contiguous context from LDS.  Staged window is
      // shifted on edge chunks; clamped columns pair with exactly-zero P.
      const int j0 = min(max(kb + shift, 0),      CTX - 8);
      const int j1 = min(max(kb + 16 + shift, 0), CTX - 8);
      v16bf b = cat8(*reinterpret_cast<const v8bf*>(vrow + j0),
                     *reinterpret_cast<const v8bf*>(vrow + j1));
      o = wmma_bf16(a, b, o);
    }
    #pragma unroll
    for (int i = 0; i < 8; ++i) {
      const int sg = qg0 + mt * 16 + laneHi * 8 + i;
      const int dd = nt * 16 + laneLo;
      out[(size_t)sg * DMODEL + h * HDIM + dd] = o[i];
    }
  }
}

// ---------------------------------------------------------------------------
extern "C" void kernel_launch(void* const* d_in, const int* in_sizes, int n_in,
                              void* d_out, int out_size, void* d_ws, size_t ws_size,
                              hipStream_t stream) {
  const float* X       = (const float*)d_in[0];
  const float* amask   = (const float*)d_in[1];
  const int*   idxmask = (const int*)  d_in[2];
  const float* Wq      = (const float*)d_in[3];
  const float* bq      = (const float*)d_in[4];
  const float* Wk      = (const float*)d_in[5];
  const float* bk      = (const float*)d_in[6];
  const float* Wv      = (const float*)d_in[7];
  const float* bv      = (const float*)d_in[8];
  float*       out     = (float*)d_out;

  // Workspace carve (all 2MB-multiple offsets, 16B aligned).
  char* w = (char*)d_ws;
  __bf16* Xb  = (__bf16*)w;  w += (size_t)S_LEN * DMODEL * 2;         // 8 MB
  __bf16* Wqb = (__bf16*)w;  w += (size_t)DMODEL * DMODEL * 2;        // 2 MB
  __bf16* Wkb = (__bf16*)w;  w += (size_t)DMODEL * DMODEL * 2;        // 2 MB
  __bf16* Wvb = (__bf16*)w;  w += (size_t)DMODEL * DMODEL * 2;        // 2 MB
  __bf16* Qb  = (__bf16*)w;  w += (size_t)NHEAD * S_LEN * HDIM * 2;   // 8 MB
  __bf16* Kb  = (__bf16*)w;  w += (size_t)NHEAD * S_LEN * HDIM * 2;   // 8 MB
  __bf16* Vtb = (__bf16*)w;  w += (size_t)NHEAD * S_LEN * HDIM * 2;   // 8 MB

  cvt_kernel <<<4096, 256, 0, stream>>>(X, Wq, Wk, Wv, Xb, Wqb, Wkb, Wvb);
  proj_kernel<<<768,  256, 0, stream>>>(Xb, Wqb, Wkb, Wvb, bq, bk, bv, Qb, Kb, Vtb);
  attn_kernel<<<2048, 256, 0, stream>>>(Qb, Kb, Vtb, amask, idxmask, out);
}
